// AdaptiveInstanceNorm_layer1_68444598829236
// MI455X (gfx1250) — matrix operand order repaired and checked by reading, not compiled, not run on hardware
//
#include <hip/hip_runtime.h>
#include <cstddef>

// ---------------------------------------------------------------------------
// CDNA5 (gfx1250, wave32) implementation of the Uformer-style reference net.
// Token GEMMs, window attention, and all 64->64 3x3 convs (implicit GEMM,
// K=576) run on V_WMMA_F32_16X16X32_F16.  GEMM A-tiles are staged to LDS via
// GLOBAL_LOAD_ASYNC_TO_LDS_B128 (builtin confirmed present on this toolchain).
// ---------------------------------------------------------------------------

typedef _Float16 h16 __attribute__((ext_vector_type(16)));
typedef float    f8  __attribute__((ext_vector_type(8)));

#define DEVINL __device__ __forceinline__

DEVINL float gelu_f(float x)  { return 0.5f * x * (1.0f + erff(x * 0.70710678118654752f)); }
DEVINL float lrelu_f(float x) { return x >= 0.f ? x : 0.01f * x; }
DEVINL float sigm_f(float x)  { return 1.0f / (1.0f + __expf(-x)); }
DEVINL float bnsc_f(float g)  { return g * rsqrtf(1.0f + 1e-5f); }   // eval BN, var=1

// ---- async global->LDS staging (CDNA5), guarded so compilation never breaks
#if defined(__has_builtin)
#if __has_builtin(__builtin_amdgcn_global_load_async_to_lds_b128)
#define ATHENA_ASYNC_LDS 1
#endif
#endif
#ifndef ATHENA_ASYNC_LDS
#define ATHENA_ASYNC_LDS 0
#endif

#if ATHENA_ASYNC_LDS
// builtin signature (from compiler diagnostic): param0 = int4 in AS1 (__device__),
// param1 = int4 in AS3 (LDS), params 2/3 = immediate offset / cpol.
typedef int v4i_ __attribute__((vector_size(16)));
typedef __attribute__((address_space(1))) v4i_ gv4i_t;
typedef __attribute__((address_space(3))) v4i_ lv4i_t;
DEVINL gv4i_t* as_global_v4(const void* p) {
  return (gv4i_t*)(unsigned long long)(size_t)p;   // flat->global: numeric identity
}
DEVINL lv4i_t* as_lds_v4(void* p) {
  return (lv4i_t*)(unsigned int)(unsigned long long)(size_t)p;  // low 32b = LDS offset
}
DEVINL void wait_async0() {
#if __has_builtin(__builtin_amdgcn_s_wait_asynccnt)
  __builtin_amdgcn_s_wait_asynccnt(0);
#else
  asm volatile("s_wait_asynccnt 0x0" ::: "memory");
#endif
}
#endif

// ---- WMMA fragment loaders (ISA 7.12.2 layouts, wave32) -------------------
// A (16x32 f16): lane m = lane&15; VGPR i holds K = (i<4?0:16)+(lane>=16?8:0)+2*(i&3)(+1)
DEVINL h16 frag_a_f32(const float* __restrict__ Arow, int k0half) {
  h16 a;
#pragma unroll
  for (int i = 0; i < 8; ++i) {
    int k = k0half + ((i < 4) ? 0 : 16) + 2 * (i & 3);
    float2 v = *reinterpret_cast<const float2*>(Arow + k);
    a[2 * i]     = (_Float16)v.x;
    a[2 * i + 1] = (_Float16)v.y;
  }
  return a;
}
DEVINL h16 frag_a_ldsh(const _Float16* __restrict__ Arow, int k0half) {
  h16 a;
#pragma unroll
  for (int i = 0; i < 8; ++i) {
    int k = k0half + ((i < 4) ? 0 : 16) + 2 * (i & 3);
    a[2 * i]     = Arow[k];
    a[2 * i + 1] = Arow[k + 1];
  }
  return a;
}
// B (32x16 f16) from W stored row-major (N x K): B[k][n] = W[n][k].
DEVINL h16 frag_b_wt(const float* __restrict__ Wrow, int kbase) {
  h16 b;
#pragma unroll
  for (int j = 0; j < 8; ++j) {
    float2 v = *reinterpret_cast<const float2*>(Wrow + kbase + ((j < 4) ? 0 : 16) + 2 * (j & 3));
    b[2 * j]     = (_Float16)v.x;
    b[2 * j + 1] = (_Float16)v.y;
  }
  return b;
}
// B from a row-major (K x N) matrix: B[k][n] = V[k*ldv + n]
DEVINL h16 frag_b_kn(const float* __restrict__ Vcol, int kbase, int ldv) {
  h16 b;
#pragma unroll
  for (int j = 0; j < 8; ++j) {
    b[2 * j]     = (_Float16)Vcol[(size_t)(kbase + ((j < 4) ? 0 : 16) + 2 * (j & 3)) * ldv];
    b[2 * j + 1] = (_Float16)Vcol[(size_t)(kbase + ((j < 4) ? 0 : 16) + 2 * (j & 3) + 1) * ldv];
  }
  return b;
}

// ---------------------------------------------------------------------------
// Token-space GEMM: Y[M,N] = act((A[M,K] x W[N,K]^T + bias) * alpha)
// block = 128 threads (4 waves) -> 16 rows x 64 cols; grid (M/16, N/64).
// The 16xK A tile (contiguous in memory) is staged once into LDS (async when
// available); all 4 waves build fragments from LDS.
// ---------------------------------------------------------------------------
__global__ __launch_bounds__(128) void gemm_tok(
    const float* __restrict__ A, const float* __restrict__ W,
    const float* __restrict__ bias, float* __restrict__ Y,
    int N, int K, float alpha, int act, int accum) {
  __shared__ float sAf[16 * 256];
  int m0 = blockIdx.x * 16;
  const float* Ablk = A + (size_t)m0 * K;   // contiguous 16*K floats
  int npkt = (16 * K) >> 2;                 // 16-byte packets
#if ATHENA_ASYNC_LDS
  for (int i = threadIdx.x; i < npkt; i += 128)
    __builtin_amdgcn_global_load_async_to_lds_b128(as_global_v4(Ablk + i * 4),
                                                   as_lds_v4(&sAf[i * 4]), 0, 0);
  wait_async0();
  __syncthreads();
#else
  for (int i = threadIdx.x; i < npkt; i += 128)
    reinterpret_cast<float4*>(sAf)[i] = reinterpret_cast<const float4*>(Ablk)[i];
  __syncthreads();
#endif
  int wave = threadIdx.x >> 5, lane = threadIdx.x & 31, half = lane >> 4;
  int n0 = (blockIdx.y * 4 + wave) * 16;
  int n = n0 + (lane & 15);
  const float* Arow = sAf + (size_t)(lane & 15) * K;
  const float* Wrow = W + (size_t)n * K;
  f8 acc = {};
  for (int k0 = 0; k0 < K; k0 += 32) {
    h16 af = frag_a_f32(Arow, k0 + half * 8);
    h16 bf = frag_b_wt(Wrow, k0 + half * 8);
    acc = __builtin_amdgcn_wmma_f32_16x16x32_f16(false, af, false, bf, (short)0, acc,
                                                 false, false);
  }
  float bv = bias ? bias[n] : 0.0f;
#pragma unroll
  for (int r = 0; r < 8; ++r) {
    int mm = m0 + r + 8 * half;
    float v = (acc[r] + bv) * alpha;
    if (act == 1) v = gelu_f(v);
    float* yp = Y + (size_t)mm * N + n;
    if (accum) *yp += v; else *yp = v;
  }
}

// ---------------------------------------------------------------------------
// 64->64 3x3 (dilated) conv as implicit GEMM on WMMA: K = 64*9 = 576.
// block = 128 threads (4 waves): 16 pixels x 64 oc; grid (2500, B).
// im2col tile built in LDS as f16; weights (oc, ic*9+tap) are already the
// needed row-major (N x K) layout.
// ---------------------------------------------------------------------------
__global__ __launch_bounds__(128) void conv3x3_wmma(
    const float* __restrict__ x, const float* __restrict__ w,
    const float* __restrict__ bias, const float* __restrict__ bng,
    const float* __restrict__ bnb, float* __restrict__ y,
    int dil, int act, int accum, int token_major) {
  __shared__ _Float16 sA[16 * 576];
  int b = blockIdx.y;
  int p0 = blockIdx.x * 16;
  for (int e = threadIdx.x; e < 16 * 576; e += 128) {
    int row = e & 15;         // pixel within tile (consecutive threads -> consecutive pixels)
    int k   = e >> 4;         // 0..575 = ic*9 + tap
    int ic = k / 9, tap = k - ic * 9;
    int p = p0 + row;
    int h = p / 200, wx = p % 200;
    int hh = h + dil * (tap / 3 - 1);
    int ww = wx + dil * (tap % 3 - 1);
    float v = 0.f;
    if ((unsigned)hh < 200u && (unsigned)ww < 200u)
      v = x[((size_t)(b * 64 + ic)) * 40000 + hh * 200 + ww];
    sA[row * 576 + k] = (_Float16)v;
  }
  __syncthreads();
  int wave = threadIdx.x >> 5, lane = threadIdx.x & 31, half = lane >> 4;
  int oc = wave * 16 + (lane & 15);
  const _Float16* Arow = sA + (size_t)(lane & 15) * 576;
  const float* Wrow = w + (size_t)oc * 576;
  f8 acc = {};
  for (int k0 = 0; k0 < 576; k0 += 32) {
    h16 af = frag_a_ldsh(Arow, k0 + half * 8);
    h16 bf = frag_b_wt(Wrow, k0 + half * 8);
    acc = __builtin_amdgcn_wmma_f32_16x16x32_f16(false, af, false, bf, (short)0, acc,
                                                 false, false);
  }
  float bv  = bias ? bias[oc] : 0.f;
  float bsc = bng ? bnsc_f(bng[oc]) : 1.f;
  float bof = bng ? bnb[oc] : 0.f;
#pragma unroll
  for (int r = 0; r < 8; ++r) {
    int p = p0 + r + 8 * half;
    float v = acc[r] + bv;
    v = v * bsc + bof;
    if (act == 3) v = lrelu_f(v);
    size_t oidx = token_major ? ((((size_t)b * 40000) + p) * 64 + oc)
                              : (((size_t)(b * 64 + oc)) * 40000 + p);
    if (accum) y[oidx] += v; else y[oidx] = v;
  }
}

// ---------------------------------------------------------------------------
// Fused window attention per (window, head): S=QK^T (+rpb, +shift mask),
// softmax in LDS, O = P V.  block = 128 threads; grid (B*NWIN, HEADS).
// ---------------------------------------------------------------------------
DEVINL int swin_cat(int h) { return (h < 192) ? 0 : (h < 196 ? 1 : 2); }

__global__ __launch_bounds__(128) void win_attn(
    const float* __restrict__ Q, const float* __restrict__ KV,
    const float* __restrict__ rpb, float* __restrict__ O, int shift) {
  __shared__ float     sS[64 * 64];
  __shared__ _Float16  sP[64 * 64];
  int widx = blockIdx.x;                 // b*625 + win
  int head = blockIdx.y;
  int win  = widx % 625;
  int wave = threadIdx.x >> 5, lane = threadIdx.x & 31, half = lane >> 4;
  size_t tok0 = (size_t)widx * 64;
  int mrow = wave * 16 + (lane & 15);

  // ---- stage 1: S = q k^T (K=32, one WMMA step per 16x16 tile) ----
  const float* Arow = Q + (tok0 + mrow) * 64 + head * 32;
  h16 af = frag_a_f32(Arow, half * 8);
  f8 acc[4];
#pragma unroll
  for (int tn = 0; tn < 4; ++tn) {
    const float* Krow = KV + (tok0 + tn * 16 + (lane & 15)) * 128 + head * 32;
    h16 bf = frag_b_wt(Krow, half * 8);
    f8 z = {};
    acc[tn] = __builtin_amdgcn_wmma_f32_16x16x32_f16(false, af, false, bf, (short)0, z,
                                                     false, false);
  }
  int wr = win / 25, wc = win % 25;
#pragma unroll
  for (int tn = 0; tn < 4; ++tn) {
#pragma unroll
    for (int r = 0; r < 8; ++r) {
      int qn = wave * 16 + r + 8 * half;       // query token
      int km = tn * 16 + (lane & 15);          // key token
      int i1 = qn >> 3, j1 = qn & 7, i2 = km >> 3, j2 = km & 7;
      float v = acc[tn][r] + rpb[((i1 - i2 + 7) * 15 + (j1 - j2 + 7)) * 2 + head];
      if (shift) {
        int r1 = swin_cat(wr * 8 + i1) * 3 + swin_cat(wc * 8 + j1);
        int r2 = swin_cat(wr * 8 + i2) * 3 + swin_cat(wc * 8 + j2);
        if (r1 != r2) v -= 100.0f;
      }
      sS[qn * 64 + km] = v;
    }
  }
  __syncthreads();

  // ---- stage 2: row softmax (threads 0..63 own one row each) ----
  if (threadIdx.x < 64) {
    float* row = sS + threadIdx.x * 64;
    float mx = -1e30f;
    for (int j = 0; j < 64; ++j) mx = fmaxf(mx, row[j]);
    float s = 0.f;
    for (int j = 0; j < 64; ++j) { float e = __expf(row[j] - mx); row[j] = e; s += e; }
    float inv = 1.0f / s;
    _Float16* pr = sP + threadIdx.x * 64;
    for (int j = 0; j < 64; ++j) pr[j] = (_Float16)(row[j] * inv);
  }
  __syncthreads();

  // ---- stage 3: O = P V  (M=64, N=32, K=64) ----
  f8 oacc[2] = {{}, {}};
  const _Float16* Prow = sP + mrow * 64;
  for (int k0 = 0; k0 < 64; k0 += 32) {
    h16 ap = frag_a_ldsh(Prow, k0 + half * 8);
#pragma unroll
    for (int tn = 0; tn < 2; ++tn) {
      const float* Vcol = KV + tok0 * 128 + 64 + head * 32 + tn * 16 + (lane & 15);
      h16 bf = frag_b_kn(Vcol, k0 + half * 8, 128);
      oacc[tn] = __builtin_amdgcn_wmma_f32_16x16x32_f16(false, ap, false, bf, (short)0,
                                                        oacc[tn], false, false);
    }
  }
#pragma unroll
  for (int tn = 0; tn < 2; ++tn) {
#pragma unroll
    for (int r = 0; r < 8; ++r) {
      int row = wave * 16 + r + 8 * half;
      int col = head * 32 + tn * 16 + (lane & 15);
      O[(tok0 + row) * 64 + col] = oacc[tn][r];
    }
  }
}

// ---------------------------------------------------------------------------
// LayerNorm over last dim (64), one wave32 per token.
// ---------------------------------------------------------------------------
__global__ __launch_bounds__(256) void ln64(const float* __restrict__ x,
                                            const float* __restrict__ g,
                                            const float* __restrict__ b,
                                            float* __restrict__ y, int ntok) {
  int wave = threadIdx.x >> 5, lane = threadIdx.x & 31;
  int tok = blockIdx.x * 8 + wave;
  if (tok >= ntok) return;
  const float* xr = x + (size_t)tok * 64;
  float v0 = xr[lane], v1 = xr[lane + 32];
  float s = v0 + v1;
#pragma unroll
  for (int off = 16; off; off >>= 1) s += __shfl_xor(s, off, 32);
  float mean = s * (1.0f / 64.0f);
  float d0 = v0 - mean, d1 = v1 - mean;
  float vs = d0 * d0 + d1 * d1;
#pragma unroll
  for (int off = 16; off; off >>= 1) vs += __shfl_xor(vs, off, 32);
  float inv = rsqrtf(vs * (1.0f / 64.0f) + 1e-5f);
  float* yr = y + (size_t)tok * 64;
  yr[lane]      = d0 * inv * g[lane]      + b[lane];
  yr[lane + 32] = d1 * inv * g[lane + 32] + b[lane + 32];
}

// ---------------------------------------------------------------------------
// Window partition gather / inverse scatter-add (handles roll by +/-shift).
// ---------------------------------------------------------------------------
__global__ __launch_bounds__(256) void window_gather(const float* __restrict__ src,
                                                     float* __restrict__ dst, int shift) {
  size_t idx = (size_t)blockIdx.x * 256 + threadIdx.x;
  if (idx >= 10240000u) return;
  int c = idx & 63;
  size_t t = idx >> 6;
  int n = t & 63;
  size_t wb = t >> 6;
  int win = wb % 625, b = wb / 625;
  int h = (win / 25) * 8 + (n >> 3) + shift;
  int w = (win % 25) * 8 + (n & 7) + shift;
  if (h >= 200) h -= 200;
  if (w >= 200) w -= 200;
  dst[idx] = src[(((size_t)b * 40000) + h * 200 + w) * 64 + c];
}

__global__ __launch_bounds__(256) void window_scatter_add(const float* __restrict__ srcw,
                                                          float* __restrict__ dstt, int shift) {
  size_t idx = (size_t)blockIdx.x * 256 + threadIdx.x;
  if (idx >= 10240000u) return;
  int c = idx & 63;
  size_t t = idx >> 6;
  int n = t & 63;
  size_t wb = t >> 6;
  int win = wb % 625, b = wb / 625;
  int h = (win / 25) * 8 + (n >> 3) + shift;
  int w = (win % 25) * 8 + (n & 7) + shift;
  if (h >= 200) h -= 200;
  if (w >= 200) w -= 200;
  dstt[(((size_t)b * 40000) + h * 200 + w) * 64 + c] += srcw[idx];
}

// ---------------------------------------------------------------------------
// NCHW 1x1 conv: each thread computes OCB output channels for its pixel
// (OCB-fold cut in input re-reads).  act: 0 none, 1 gelu(+BN), 2 relu.
// grid (ceil(HW/256), B*OC/OCB).
// ---------------------------------------------------------------------------
template <int OCB>
__global__ __launch_bounds__(256) void conv1x1k(const float* __restrict__ x,
                                                const float* __restrict__ w,
                                                const float* __restrict__ bias,
                                                const float* __restrict__ bng,
                                                const float* __restrict__ bnb,
                                                const float* __restrict__ style,
                                                float* __restrict__ y,
                                                int IC, int OC, int HW, int act) {
  int p = blockIdx.x * 256 + threadIdx.x;
  if (p >= HW) return;
  int ocg = OC / OCB;
  int b = blockIdx.y / ocg, oc0 = (blockIdx.y % ocg) * OCB;
  const float* xb = x + (size_t)b * IC * HW + p;
  float acc[OCB];
#pragma unroll
  for (int j = 0; j < OCB; ++j) acc[j] = 0.f;
  for (int ic = 0; ic < IC; ++ic) {
    float xv = xb[(size_t)ic * HW];
    if (style) xv *= style[b * IC + ic];
#pragma unroll
    for (int j = 0; j < OCB; ++j) acc[j] += w[(oc0 + j) * IC + ic] * xv;
  }
#pragma unroll
  for (int j = 0; j < OCB; ++j) {
    int oc = oc0 + j;
    float a = acc[j];
    if (bias) a += bias[oc];
    if (act == 1) a = gelu_f(a);
    else if (act == 2) a = fmaxf(a, 0.f);
    if (bng) a = a * bnsc_f(bng[oc]) + bnb[oc];
    y[((size_t)(b * OC + oc)) * HW + p] = a;
  }
}

// ---------------------------------------------------------------------------
// LeFF depthwise 3x3 + GELU on token-major hidden (B,HW,256).
// ---------------------------------------------------------------------------
__global__ __launch_bounds__(256) void dwconv3x3_gelu(const float* __restrict__ x,
                                                      const float* __restrict__ w,
                                                      const float* __restrict__ bias,
                                                      float* __restrict__ y) {
  size_t idx = (size_t)blockIdx.x * 256 + threadIdx.x;
  if (idx >= 40960000u) return;
  int c = idx & 255;
  size_t l = idx >> 8;
  int pos = l % 40000, b = l / 40000;
  int h = pos / 200, wx = pos % 200;
  float a = bias[c];
#pragma unroll
  for (int ky = 0; ky < 3; ++ky) {
    int hh = h + ky - 1;
    if ((unsigned)hh >= 200u) continue;
#pragma unroll
    for (int kx = 0; kx < 3; ++kx) {
      int ww = wx + kx - 1;
      if ((unsigned)ww >= 200u) continue;
      a += w[c * 9 + ky * 3 + kx] * x[(((size_t)b * 40000) + hh * 200 + ww) * 256 + c];
    }
  }
  y[idx] = gelu_f(a);
}

// ---------------------------------------------------------------------------
// Reductions + small fused kernels (style / CBAM / head).
// ---------------------------------------------------------------------------
__global__ __launch_bounds__(256) void row_mean(const float* __restrict__ x,
                                                float* __restrict__ y, int HW, float scale) {
  __shared__ float sm[256];
  int row = blockIdx.x;
  float s = 0.f;
  for (int i = threadIdx.x; i < HW; i += 256) s += x[(size_t)row * HW + i];
  sm[threadIdx.x] = s; __syncthreads();
  for (int o = 128; o; o >>= 1) { if (threadIdx.x < o) sm[threadIdx.x] += sm[threadIdx.x + o]; __syncthreads(); }
  if (threadIdx.x == 0) y[row] = sm[0] * scale;
}

__global__ __launch_bounds__(256) void row_mean_max(const float* __restrict__ x,
                                                    float* __restrict__ ym, float* __restrict__ yx,
                                                    int HW) {
  __shared__ float sm[256], sx[256];
  int row = blockIdx.x;
  float s = 0.f, m = -1e30f;
  for (int i = threadIdx.x; i < HW; i += 256) { float v = x[(size_t)row * HW + i]; s += v; m = fmaxf(m, v); }
  sm[threadIdx.x] = s; sx[threadIdx.x] = m; __syncthreads();
  for (int o = 128; o; o >>= 1) {
    if (threadIdx.x < o) { sm[threadIdx.x] += sm[threadIdx.x + o]; sx[threadIdx.x] = fmaxf(sx[threadIdx.x], sx[threadIdx.x + o]); }
    __syncthreads();
  }
  if (threadIdx.x == 0) { ym[row] = sm[0] / HW; yx[row] = sx[0]; }
}

__global__ __launch_bounds__(128) void style_mlp(const float* __restrict__ mean,
                                                 float* __restrict__ style,
                                                 const float* w0, const float* w1, const float* w2,
                                                 const float* w3, const float* w4, const float* w5,
                                                 const float* b0, const float* b1, const float* b2,
                                                 const float* b3, const float* b4, const float* b5) {
  __shared__ float sin_[128];
  int b = blockIdx.x, c = threadIdx.x;
  const float* ws[6] = {w0, w1, w2, w3, w4, w5};
  const float* bs[6] = {b0, b1, b2, b3, b4, b5};
  sin_[c] = mean[b * 128 + c];
  __syncthreads();
  const float scale = 0.125f;  // sqrt(2/128)
  for (int l = 0; l < 6; ++l) {
    float a = 0.f;
    for (int i = 0; i < 128; ++i) a += sin_[i] * scale * ws[l][c * 128 + i];
    a = fmaxf(a + bs[l][c], 0.f);
    __syncthreads();
    sin_[c] = a;
    __syncthreads();
  }
  style[b * 128 + c] = sin_[c];
}

__global__ __launch_bounds__(64) void ca_gate(const float* __restrict__ avg,
                                              const float* __restrict__ mx,
                                              const float* __restrict__ w1,
                                              const float* __restrict__ w2,
                                              float* __restrict__ cas) {
  __shared__ float ha[4], hm[4];
  int b = blockIdx.x, c = threadIdx.x;
  if (c < 4) {
    float sa = 0.f, sm = 0.f;
    for (int i = 0; i < 64; ++i) { sa += w1[c * 64 + i] * avg[b * 64 + i]; sm += w1[c * 64 + i] * mx[b * 64 + i]; }
    ha[c] = fmaxf(sa, 0.f); hm[c] = fmaxf(sm, 0.f);
  }
  __syncthreads();
  float o = 0.f;
  for (int j = 0; j < 4; ++j) o += w2[c * 4 + j] * (ha[j] + hm[j]);
  cas[b * 64 + c] = sigm_f(o);
}

__global__ __launch_bounds__(256) void apply_ca(float* __restrict__ out,
                                                const float* __restrict__ cas) {
  size_t idx = (size_t)blockIdx.x * 256 + threadIdx.x;
  if (idx >= 10240000u) return;
  int bc = idx / 40000;
  out[idx] *= cas[bc];
}

__global__ __launch_bounds__(256) void chan_stats(const float* __restrict__ x,
                                                  float* __restrict__ mean,
                                                  float* __restrict__ mx) {
  int p = blockIdx.x * 256 + threadIdx.x;
  if (p >= 40000) return;
  int b = blockIdx.y;
  const float* xb = x + (size_t)b * 64 * 40000 + p;
  float s = 0.f, m = -1e30f;
  for (int c = 0; c < 64; ++c) { float v = xb[(size_t)c * 40000]; s += v; m = fmaxf(m, v); }
  mean[b * 40000 + p] = s * (1.0f / 64.0f);
  mx[b * 40000 + p] = m;
}

__global__ __launch_bounds__(256) void sa_apply(float* __restrict__ out,
                                                const float* __restrict__ mean,
                                                const float* __restrict__ mx,
                                                const float* __restrict__ sw) {
  int p = blockIdx.x * 256 + threadIdx.x;
  if (p >= 40000) return;
  int b = blockIdx.y;
  int h = p / 200, wx = p % 200;
  float a = 0.f;
#pragma unroll
  for (int ky = 0; ky < 7; ++ky) {
    int hh = h + ky - 3;
    if ((unsigned)hh >= 200u) continue;
#pragma unroll
    for (int kx = 0; kx < 7; ++kx) {
      int ww = wx + kx - 3;
      if ((unsigned)ww >= 200u) continue;
      int q = hh * 200 + ww;
      a += sw[ky * 7 + kx] * mean[b * 40000 + q] + sw[49 + ky * 7 + kx] * mx[b * 40000 + q];
    }
  }
  float s = sigm_f(a);
  for (int c = 0; c < 64; ++c) out[((size_t)(b * 64 + c)) * 40000 + p] *= s;
}

__global__ __launch_bounds__(256) void add2(const float* __restrict__ a,
                                            const float* __restrict__ b,
                                            float* __restrict__ y, size_t n) {
  size_t idx = (size_t)blockIdx.x * 256 + threadIdx.x;
  if (idx < n) y[idx] = a[idx] + b[idx];
}

// ConvTranspose2d(16->4, k=2, s=2): y4 (B,4,400,400)
__global__ __launch_bounds__(256) void up_tconv(const float* __restrict__ rt,
                                                const float* __restrict__ w,
                                                const float* __restrict__ bias,
                                                float* __restrict__ y) {
  size_t idx = (size_t)blockIdx.x * 256 + threadIdx.x;
  if (idx >= 2560000u) return;
  int pos = idx % 160000;
  int o = (idx / 160000) % 4;
  int b = idx / 640000;
  int yy = pos / 400, xx = pos % 400;
  int h = yy >> 1, u = yy & 1, ww = xx >> 1, v = xx & 1;
  float a = bias[o];
  for (int c = 0; c < 16; ++c)
    a += rt[((size_t)(b * 16 + c)) * 40000 + h * 200 + ww] * w[((c * 4 + o) * 2 + u) * 2 + v];
  y[idx] = a;
}

// ---------------------------------------------------------------------------
// Orchestration
// ---------------------------------------------------------------------------
extern "C" void kernel_launch(void* const* d_in, const int* in_sizes, int n_in,
                              void* d_out, int out_size, void* d_ws, size_t ws_size,
                              hipStream_t stream) {
  (void)in_sizes; (void)n_in; (void)out_size; (void)ws_size;
  auto F = [&](int i) { return (const float*)d_in[i]; };
  const float* x = F(0);

  float* ws   = (float*)d_ws;
  float* T    = ws;                    // tokens (B,HW,64)       41 MB
  float* HWb  = ws + 10240000;         // windowed / attn-out    41 MB (also y4 head buf)
  float* Qb   = ws + 20480000;         // q / proj-out           41 MB
  float* KVb  = ws + 30720000;         // kv                     82 MB
  float* LNb  = ws + 51200000;         // ln out / rt out        41 MB
  float* HID  = ws + 61440000;         // LeFF hidden           164 MB (aliases TMP128)
  float* HID2 = ws + 102400000;        // LeFF hidden post-dw   164 MB
  float* OUTA = ws + 143360000;        // (B,64,200,200)         41 MB
  float* OUTS = ws + 153600000;        // short-structure sum    41 MB
  float* SM   = ws + 163840000;        // small stats
  float* styleMean = SM;               // 512
  float* styleVec  = SM + 512;         // 512
  float* caAvg = SM + 1024;            // 256
  float* caMax = SM + 1280;            // 256
  float* caS   = SM + 1536;            // 256
  float* spMean = SM + 2048;           // 160000
  float* spMax  = SM + 162048;         // 160000
  float* TMP128 = HID;                 // phase-A alias (B,128,200,200)
  float* Y4 = HWb;                     // phase-D alias (B,4,400,400)
  float* RT = LNb;                     // phase-D alias (B,16,200,200)

  dim3 b256(256);

  // ---- Phase A: style vector -------------------------------------------
  conv1x1k<8><<<dim3(157, 4 * 16), b256, 0, stream>>>(x, F(1), F(2), F(3), F(4), nullptr,
                                                      TMP128, 128, 128, 40000, 1);
  row_mean<<<512, b256, 0, stream>>>(TMP128, styleMean, 40000, 1.0f / 40000.0f);
  style_mlp<<<4, 128, 0, stream>>>(styleMean, styleVec,
                                   F(5), F(6), F(7), F(8), F(9), F(10),
                                   F(11), F(12), F(13), F(14), F(15), F(16));

  // ---- Phase B: resize (style-gated), short structure, input proj ------
  conv1x1k<8><<<dim3(157, 4 * 8), b256, 0, stream>>>(x, F(17), F(18), F(19), F(20),
                                                     styleVec, OUTA, 128, 64, 40000, 1);
  conv3x3_wmma<<<dim3(2500, 4), 128, 0, stream>>>(OUTA, F(21), nullptr, F(22), F(23), OUTS, 1, 3, 0, 0);
  conv3x3_wmma<<<dim3(2500, 4), 128, 0, stream>>>(OUTA, F(24), nullptr, F(25), F(26), OUTS, 3, 3, 1, 0);
  conv3x3_wmma<<<dim3(2500, 4), 128, 0, stream>>>(OUTA, F(27), nullptr, F(28), F(29), OUTS, 5, 3, 1, 0);
  conv3x3_wmma<<<dim3(2500, 4), 128, 0, stream>>>(OUTA, F(30), F(31), nullptr, nullptr, T, 1, 3, 0, 1);

  // ---- Phase C: 3 LeWin blocks (WMMA GEMMs + fused window attention) ---
  for (int blk = 0; blk < 3; ++blk) {
    int p0 = 32 + 17 * blk;
    int shift = (blk & 1) ? 4 : 0;
    ln64<<<20000, b256, 0, stream>>>(T, F(p0 + 0), F(p0 + 1), LNb, 160000);
    window_gather<<<40000, b256, 0, stream>>>(LNb, HWb, shift);
    gemm_tok<<<dim3(10000, 1), 128, 0, stream>>>(HWb, F(p0 + 2), F(p0 + 3), Qb,
                                                 64, 64, 0.17677669529663688f, 0, 0);
    gemm_tok<<<dim3(10000, 2), 128, 0, stream>>>(HWb, F(p0 + 4), F(p0 + 5), KVb,
                                                 128, 64, 1.0f, 0, 0);
    win_attn<<<dim3(2500, 2), 128, 0, stream>>>(Qb, KVb, F(p0 + 8), HWb, shift);
    gemm_tok<<<dim3(10000, 1), 128, 0, stream>>>(HWb, F(p0 + 6), F(p0 + 7), Qb,
                                                 64, 64, 1.0f, 0, 0);
    window_scatter_add<<<40000, b256, 0, stream>>>(Qb, T, shift);
    ln64<<<20000, b256, 0, stream>>>(T, F(p0 + 9), F(p0 + 10), LNb, 160000);
    gemm_tok<<<dim3(10000, 4), 128, 0, stream>>>(LNb, F(p0 + 11), F(p0 + 12), HID,
                                                 256, 64, 1.0f, 1, 0);
    dwconv3x3_gelu<<<160000, b256, 0, stream>>>(HID, F(p0 + 13), F(p0 + 14), HID2);
    gemm_tok<<<dim3(10000, 1), 128, 0, stream>>>(HID2, F(p0 + 15), F(p0 + 16), T,
                                                 64, 256, 1.0f, 0, 1);
  }

  // ---- Phase D: merge, CBAM, head --------------------------------------
  add2<<<40000, b256, 0, stream>>>(T, OUTS, OUTA, 10240000u);  // torch .reshape is a raw view
  row_mean_max<<<256, b256, 0, stream>>>(OUTA, caAvg, caMax, 40000);
  ca_gate<<<4, 64, 0, stream>>>(caAvg, caMax, F(83), F(84), caS);
  apply_ca<<<40000, b256, 0, stream>>>(OUTA, caS);
  chan_stats<<<dim3(157, 4), b256, 0, stream>>>(OUTA, spMean, spMax);
  sa_apply<<<dim3(157, 4), b256, 0, stream>>>(OUTA, spMean, spMax, F(85));
  conv1x1k<8><<<dim3(157, 4 * 2), b256, 0, stream>>>(OUTA, F(86), F(87), F(88), F(89),
                                                     nullptr, RT, 64, 16, 40000, 1);
  up_tconv<<<10000, b256, 0, stream>>>(RT, F(90), F(91), Y4);
  conv1x1k<5><<<dim3(625, 4), b256, 0, stream>>>(Y4, F(92), nullptr, nullptr, nullptr,
                                                 nullptr, (float*)d_out, 4, 5, 160000, 2);
}
